// AGDNMotionForecaster_79946521247953
// MI455X (gfx1250) — matrix operand order repaired and checked by reading, not compiled, not run on hardware
//
#include <hip/hip_runtime.h>
#include <hip/hip_bf16.h>
#include <math.h>

typedef __attribute__((ext_vector_type(16))) _Float16 v16h;
typedef __attribute__((ext_vector_type(8)))  float    v8f;

#define B_   16
#define L_   128
#define DIN  8
#define H_   128
#define NL_  2
#define FUT_ 12
#define MD_  2
#define IT_  8        // i-rows per block in edge kernel
#define MPAIR 128     // IT_*16 pair-rows per j-tile

// ---- WMMA fragment helpers (contiguous k order: element e of lane half h holds k = 16h + e, same for A and B) ----
__device__ __forceinline__ v16h frag_ld_a(const _Float16* p, int stride, int lane) {
  // A 16x32 f16: lane<16 -> row=lane, K = 0..15 ; lane>=16 -> K = 16..31
  const _Float16* r = p + (lane & 15) * stride + ((lane >> 4) << 4);
  v16h f;
#pragma unroll
  for (int e = 0; e < 16; ++e) f[e] = r[e];
  return f;
}
__device__ __forceinline__ v16h frag_ld_b(const _Float16* p, int stride, int lane) {
  // B 32x16 f16 from transposed weight [N][K]: lane&15 -> col, (lane>>4)*16 + 0..15 -> K
  const _Float16* r = p + (lane & 15) * stride + ((lane >> 4) << 4);
  v16h f;
#pragma unroll
  for (int e = 0; e < 16; ++e) f[e] = r[e];
  return f;
}
__device__ __forceinline__ v8f wmma16(v16h a, v16h b, v8f c) {
  return __builtin_amdgcn_wmma_f32_16x16x32_f16(false, a, false, b, (short)0, c, false, false);
}

// ---- prep: cumsum of max(history[...,5],0) per batch ----
__global__ void k_ts(const float* __restrict__ hist, float* __restrict__ ts) {
  int b = threadIdx.x;
  if (b >= B_) return;
  float acc = 0.f;
  for (int l = 0; l < L_; ++l) {
    float v = hist[(b * L_ + l) * DIN + 5];
    acc += fmaxf(v, 0.f);
    ts[b * L_ + l] = acc;
  }
}

// ---- prep: h0 = (history @ Wp + bp) * vm ; also f16 copy ----
__global__ void k_h0(const float* __restrict__ hist, const float* __restrict__ mask,
                     const float* __restrict__ Wp, const float* __restrict__ bp,
                     float* __restrict__ h, _Float16* __restrict__ h16) {
  int row = blockIdx.x;           // b*L + l
  int n = threadIdx.x;
  float acc = bp[n];
#pragma unroll
  for (int c = 0; c < DIN; ++c) acc = fmaf(hist[row * DIN + c], Wp[c * H_ + n], acc);
  float vm = (mask[row] > 0.f) ? 1.f : 0.f;
  float v = acc * vm;
  h[row * H_ + n] = v;
  h16[row * H_ + n] = (_Float16)v;
}

// ---- prep: transpose+convert all layer weights to f16 [N][K] ----
// per-layer f16 layout (halfs): Wcat_t[512][128] @0, We2t @65536, Wa1_2t @81920,
// Wm1t @98304, Wm2t @114688, Wo1t2 @131072, Wo2t @147456 ; layer stride 163840
__global__ void k_wconv(const float* __restrict__ Wa1, const float* __restrict__ Wm1,
                        const float* __restrict__ Wo1, const float* __restrict__ We2,
                        const float* __restrict__ Wm2, const float* __restrict__ Wo2,
                        _Float16* __restrict__ wf) {
  int gid = blockIdx.x * blockDim.x + threadIdx.x;
  if (gid >= NL_ * 163840) return;
  int l = gid / 163840;
  int r = gid % 163840;
  _Float16* dst = wf + l * 163840;
  float v;
  if (r < 65536) {                       // Wcat_t: [sa|da|sm|so] x [K]
    int n = r >> 7, k = r & 127;
    int np = n & 127, g = n >> 7;
    if (g == 0)      v = Wa1[(l * 384 + k) * H_ + np];
    else if (g == 1) v = Wa1[(l * 384 + 128 + k) * H_ + np];
    else if (g == 2) v = Wm1[(l * 256 + k) * H_ + np];
    else             v = Wo1[(l * 256 + k) * H_ + np];
  } else {
    int q = r - 65536;
    int w = q >> 14, idx = q & 16383;
    int n = idx >> 7, k = idx & 127;
    switch (w) {
      case 0:  v = We2[(l * 128 + k) * H_ + n]; break;
      case 1:  v = Wa1[(l * 384 + 256 + k) * H_ + n]; break;
      case 2:  v = Wm1[(l * 256 + 128 + k) * H_ + n]; break;
      case 3:  v = Wm2[(l * 128 + k) * H_ + n]; break;
      case 4:  v = Wo1[(l * 256 + 128 + k) * H_ + n]; break;
      default: v = Wo2[(l * 128 + k) * H_ + n]; break;
    }
  }
  dst[r] = (_Float16)v;
}

// ---- per layer: proj = h @ [Wa1[:H] | Wa1[H:2H] | Wm1[:H] | Wo1[:H]]  (WMMA) ----
__global__ void k_proj(const _Float16* __restrict__ h16, const _Float16* __restrict__ wcat,
                       float* __restrict__ proj) {
  int mt = blockIdx.x;            // 128 M-tiles of 16 rows over B*L=2048
  int lane = threadIdx.x & 31;
  int wv = threadIdx.x >> 5;      // 4 waves
  const _Float16* as = h16 + (mt * 16) * H_;
  v16h a0 = frag_ld_a(as,      H_, lane);
  v16h a1 = frag_ld_a(as + 32, H_, lane);
  v16h a2 = frag_ld_a(as + 64, H_, lane);
  v16h a3 = frag_ld_a(as + 96, H_, lane);
  for (int q = 0; q < 8; ++q) {
    int nt = wv * 8 + q;          // 32 N-tiles over 512
    const _Float16* wb = wcat + (nt * 16) * H_;
    v8f acc = {};
    acc = wmma16(a0, frag_ld_b(wb,      H_, lane), acc);
    acc = wmma16(a1, frag_ld_b(wb + 32, H_, lane), acc);
    acc = wmma16(a2, frag_ld_b(wb + 64, H_, lane), acc);
    acc = wmma16(a3, frag_ld_b(wb + 96, H_, lane), acc);
    int col = nt * 16 + (lane & 15);
    int rbase = mt * 16 + ((lane >> 4) << 3);
#pragma unroll
    for (int r = 0; r < 8; ++r) proj[(rbase + r) * 512 + col] = acc[r];
  }
}

// ---- fused edge/attention/message layer kernel ----
__global__ void k_edge(const float* __restrict__ hist, const float* __restrict__ mask,
                       const float* __restrict__ ts, float* __restrict__ h,
                       _Float16* __restrict__ h16, const float* __restrict__ proj,
                       const _Float16* __restrict__ wf,
                       const float* __restrict__ We1, const float* __restrict__ be1,
                       const float* __restrict__ be2, const float* __restrict__ ba1,
                       const float* __restrict__ wa2, const float* __restrict__ ba2,
                       const float* __restrict__ bm1, const float* __restrict__ bm2,
                       const float* __restrict__ bo1, const float* __restrict__ bo2,
                       const float* __restrict__ lng, const float* __restrict__ lnb,
                       int l) {
  int tid = threadIdx.x;
  int lane = tid & 31;
  int wv = tid >> 5;                       // 8 waves
  int b = blockIdx.x >> 4;
  int i0 = (blockIdx.x & 15) * IT_;

  extern __shared__ char smem[];
  _Float16* sA   = (_Float16*)(smem);              // [128][128] f16
  _Float16* sEh  = (_Float16*)(smem + 32768);      // [128][128] f16
  float* sAgg    = (float*)(smem + 65536);         // [8][128]
  float* sHi     = (float*)(smem + 69632);         // [8][128]
  float* sWe1    = (float*)(smem + 73728);         // [4][128]
  float* sBe1    = (float*)(smem + 75776);         // [128]
  float* sJlat   = (float*)(smem + 76288);
  float* sJlon   = (float*)(smem + 76800);
  float* sJts    = (float*)(smem + 77312);
  float* sJsrc   = (float*)(smem + 77824);
  float* sJmask  = (float*)(smem + 78336);
  float* sAtt    = (float*)(smem + 78848);         // [128]
  float* sWgt    = (float*)(smem + 79360);         // [128]
  float* sRowM   = (float*)(smem + 79872);
  float* sRowS   = (float*)(smem + 79904);
  float* sScale  = (float*)(smem + 79936);
  float* sMu     = (float*)(smem + 79968);
  float* sVar    = (float*)(smem + 80000);

  const _Float16* wbase = wf + l * 163840;
  const _Float16* We2t  = wbase + 65536;
  const _Float16* Wa12t = wbase + 81920;
  const _Float16* Wm1t  = wbase + 98304;
  const _Float16* Wm2t  = wbase + 114688;
  const _Float16* Wo1t2 = wbase + 131072;
  const _Float16* Wo2t  = wbase + 147456;
  const float invs = 0.08838834764831843f;     // 1/sqrt(H)

  // prefetch the 6 stage-weight matrices (192 KB f16) toward the caches
  for (int x = tid; x < 1536; x += 256)
    __builtin_prefetch(We2t + x * 64, 0, 1);

  for (int j = tid; j < L_; j += 256) {
    const float* hr = hist + (b * L_ + j) * DIN;
    sJlat[j] = hr[0]; sJlon[j] = hr[1]; sJsrc[j] = hr[6];
    sJts[j] = ts[b * L_ + j]; sJmask[j] = mask[b * L_ + j];
  }
  for (int x = tid; x < 4 * H_; x += 256) sWe1[x] = We1[l * 4 * H_ + x];
  for (int x = tid; x < H_; x += 256) sBe1[x] = be1[l * H_ + x];
  for (int x = tid; x < IT_ * H_; x += 256) {
    sHi[x] = h[(b * L_ + i0 + (x >> 7)) * H_ + (x & 127)];
    sAgg[x] = 0.f;
  }
  if (tid < IT_) { sRowM[tid] = -3.0e38f; sRowS[tid] = 0.f; }
  __syncthreads();

  for (int jt = 0; jt < L_ / 16; ++jt) {
    int j0 = jt * 16;
    // 1) edge features + t1 = relu(ef@We1+be1) -> sA (f16); all 256 threads,
    //    thread halves split the N range of pair p = tid&127
    {
      int p = tid & 127;
      int nh = tid >> 7;
      int il = p >> 4, jl = p & 15;
      int i = i0 + il, j = j0 + jl;
      float dlat = sJlat[i] - sJlat[j], dlon = sJlon[i] - sJlon[j];
      float ef0 = sqrtf(dlat * dlat + dlon * dlon + 1e-8f);
      float ef1 = fabsf(sJts[i] - sJts[j]) * (1.f / 300.f);
      float ef2 = (sJsrc[i] == sJsrc[j]) ? 1.f : 0.f;
      float ef3 = (i == j) ? 1.f : 0.f;
      int n0 = nh * 64;
      _Float16* dst = sA + p * H_ + n0;
      for (int n = 0; n < 64; ++n) {
        int nn = n0 + n;
        float v = fmaf(ef0, sWe1[nn], fmaf(ef1, sWe1[H_ + nn],
                  fmaf(ef2, sWe1[2 * H_ + nn], fmaf(ef3, sWe1[3 * H_ + nn], sBe1[nn]))));
        dst[n] = (_Float16)fmaxf(v, 0.f);
      }
    }
    __syncthreads();
    // 2) eh = t1 @ We2 + be2 -> sEh (f16)       [WMMA, B hoisted]
    {
      const _Float16* wb = We2t + (wv * 16) * H_;
      v16h b0 = frag_ld_b(wb,      H_, lane);
      v16h b1 = frag_ld_b(wb + 32, H_, lane);
      v16h b2 = frag_ld_b(wb + 64, H_, lane);
      v16h b3 = frag_ld_b(wb + 96, H_, lane);
      int n = wv * 16 + (lane & 15);
      float bias = be2[l * H_ + n];
      for (int mt = 0; mt < 8; ++mt) {
        const _Float16* as = sA + mt * 16 * H_;
        v8f acc = {};
        acc = wmma16(frag_ld_a(as,      H_, lane), b0, acc);
        acc = wmma16(frag_ld_a(as + 32, H_, lane), b1, acc);
        acc = wmma16(frag_ld_a(as + 64, H_, lane), b2, acc);
        acc = wmma16(frag_ld_a(as + 96, H_, lane), b3, acc);
        int rbase = mt * 16 + ((lane >> 4) << 3);
#pragma unroll
        for (int r = 0; r < 8; ++r) sEh[(rbase + r) * H_ + n] = (_Float16)(acc[r] + bias);
      }
    }
    __syncthreads();
    if (tid < MPAIR) sAtt[tid] = 0.f;
    __syncthreads();
    // 3) att logits: tanh(sa+da+eh@Wa1_2+ba1) . wa2   [WMMA + shuffle-tree reduce]
    {
      const _Float16* wb = Wa12t + (wv * 16) * H_;
      v16h b0 = frag_ld_b(wb,      H_, lane);
      v16h b1 = frag_ld_b(wb + 32, H_, lane);
      v16h b2 = frag_ld_b(wb + 64, H_, lane);
      v16h b3 = frag_ld_b(wb + 96, H_, lane);
      int n = wv * 16 + (lane & 15);
      int hi = lane >> 4;
      float w2 = wa2[l * H_ + n];
      float ba = ba1[l * H_ + n];
      for (int mt = 0; mt < 8; ++mt) {
        const _Float16* as = sEh + mt * 16 * H_;
        v8f acc = {};
        acc = wmma16(frag_ld_a(as,      H_, lane), b0, acc);
        acc = wmma16(frag_ld_a(as + 32, H_, lane), b1, acc);
        acc = wmma16(frag_ld_a(as + 64, H_, lane), b2, acc);
        acc = wmma16(frag_ld_a(as + 96, H_, lane), b3, acc);
        int i = i0 + mt;
        float sa_ = proj[(b * L_ + i) * 512 + n];
#pragma unroll
        for (int r = 0; r < 8; ++r) {
          int jl = r + (hi << 3);
          float da_ = proj[(b * L_ + j0 + jl) * 512 + H_ + n];
          float v = tanhf(acc[r] + sa_ + da_ + ba) * w2;
          v += __shfl_xor(v, 1, 32);
          v += __shfl_xor(v, 2, 32);
          v += __shfl_xor(v, 4, 32);
          v += __shfl_xor(v, 8, 32);
          if ((lane & 15) == 0) atomicAdd(&sAtt[mt * 16 + jl], v);
        }
      }
    }
    __syncthreads();
    // 4) online softmax per i-row
    if (tid < IT_) {
      int i = i0 + tid;
      int qbad = (sJmask[i] <= 0.f);
      float m_old = sRowM[tid], tm = -3.0e38f;
      float av[16];
      for (int jl = 0; jl < 16; ++jl) {
        float a = (sAtt[tid * 16 + jl] + ba2[l]) * invs;
        if (qbad || sJmask[j0 + jl] <= 0.f) a = -1e9f;
        av[jl] = a; tm = fmaxf(tm, a);
      }
      float m_new = fmaxf(m_old, tm);
      float sc = __expf(m_old - m_new);
      float s = sRowS[tid] * sc;
      for (int jl = 0; jl < 16; ++jl) {
        float e = __expf(av[jl] - m_new);
        sWgt[tid * 16 + jl] = e; s += e;
      }
      sRowM[tid] = m_new; sRowS[tid] = s; sScale[tid] = sc;
    }
    __syncthreads();
    for (int x = tid; x < IT_ * H_; x += 256) sAgg[x] *= sScale[x >> 7];
    // 5) msg hidden = relu(sm + eh@Wm1_2 + bm1) -> sA (f16)   [WMMA, B hoisted]
    {
      const _Float16* wb = Wm1t + (wv * 16) * H_;
      v16h b0 = frag_ld_b(wb,      H_, lane);
      v16h b1 = frag_ld_b(wb + 32, H_, lane);
      v16h b2 = frag_ld_b(wb + 64, H_, lane);
      v16h b3 = frag_ld_b(wb + 96, H_, lane);
      int n = wv * 16 + (lane & 15);
      float bm = bm1[l * H_ + n];
      for (int mt = 0; mt < 8; ++mt) {
        const _Float16* as = sEh + mt * 16 * H_;
        v8f acc = {};
        acc = wmma16(frag_ld_a(as,      H_, lane), b0, acc);
        acc = wmma16(frag_ld_a(as + 32, H_, lane), b1, acc);
        acc = wmma16(frag_ld_a(as + 64, H_, lane), b2, acc);
        acc = wmma16(frag_ld_a(as + 96, H_, lane), b3, acc);
        float sm_ = proj[(b * L_ + i0 + mt) * 512 + 2 * H_ + n];
        int rbase = mt * 16 + ((lane >> 4) << 3);
#pragma unroll
        for (int r = 0; r < 8; ++r)
          sA[(rbase + r) * H_ + n] = (_Float16)fmaxf(acc[r] + sm_ + bm, 0.f);
      }
    }
    __syncthreads();
    // 6) msg = mh @ Wm2 + bm2 ; agg += wgt * msg   [WMMA + shuffle reduce]
    {
      const _Float16* wb = Wm2t + (wv * 16) * H_;
      v16h b0 = frag_ld_b(wb,      H_, lane);
      v16h b1 = frag_ld_b(wb + 32, H_, lane);
      v16h b2 = frag_ld_b(wb + 64, H_, lane);
      v16h b3 = frag_ld_b(wb + 96, H_, lane);
      int n = wv * 16 + (lane & 15);
      float bm = bm2[l * H_ + n];
      int hi = lane >> 4;
      for (int mt = 0; mt < 8; ++mt) {
        const _Float16* as = sA + mt * 16 * H_;
        v8f acc = {};
        acc = wmma16(frag_ld_a(as,      H_, lane), b0, acc);
        acc = wmma16(frag_ld_a(as + 32, H_, lane), b1, acc);
        acc = wmma16(frag_ld_a(as + 64, H_, lane), b2, acc);
        acc = wmma16(frag_ld_a(as + 96, H_, lane), b3, acc);
        float part = 0.f;
#pragma unroll
        for (int r = 0; r < 8; ++r) {
          int jl = r + (hi << 3);
          part += sWgt[mt * 16 + jl] * (acc[r] + bm);
        }
        part += __shfl_xor(part, 16, 32);
        if (lane < 16) sAgg[mt * H_ + n] += part;
      }
    }
    __syncthreads();
  }

  // normalize agg, pack [16][128] f16 (rows 8..15 zero) into sA
  for (int x = tid; x < 16 * H_; x += 256) {
    int il = x >> 7;
    float v = (il < IT_) ? sAgg[x] / sRowS[il] : 0.f;
    sA[x] = (_Float16)v;
  }
  __syncthreads();
  // out1 = relu(so + agg@Wo1_2 + bo1) -> sEh rows 0..15   [WMMA]
  {
    const _Float16* wb = Wo1t2 + (wv * 16) * H_;
    v8f acc = {};
    acc = wmma16(frag_ld_a(sA,      H_, lane), frag_ld_b(wb,      H_, lane), acc);
    acc = wmma16(frag_ld_a(sA + 32, H_, lane), frag_ld_b(wb + 32, H_, lane), acc);
    acc = wmma16(frag_ld_a(sA + 64, H_, lane), frag_ld_b(wb + 64, H_, lane), acc);
    acc = wmma16(frag_ld_a(sA + 96, H_, lane), frag_ld_b(wb + 96, H_, lane), acc);
    int n = wv * 16 + (lane & 15);
    int hi = lane >> 4;
    float bo = bo1[l * H_ + n];
#pragma unroll
    for (int r = 0; r < 8; ++r) {
      int il = r + (hi << 3);
      float v = 0.f;
      if (il < IT_) {
        float so_ = proj[(b * L_ + i0 + il) * 512 + 3 * H_ + n];
        v = fmaxf(acc[r] + so_ + bo, 0.f);
      }
      sEh[il * H_ + n] = (_Float16)v;
    }
  }
  __syncthreads();
  // out2 = out1@Wo2 + bo2 ; x = h_old + out2 -> sAgg   [WMMA]
  {
    const _Float16* wb = Wo2t + (wv * 16) * H_;
    v8f acc = {};
    acc = wmma16(frag_ld_a(sEh,      H_, lane), frag_ld_b(wb,      H_, lane), acc);
    acc = wmma16(frag_ld_a(sEh + 32, H_, lane), frag_ld_b(wb + 32, H_, lane), acc);
    acc = wmma16(frag_ld_a(sEh + 64, H_, lane), frag_ld_b(wb + 64, H_, lane), acc);
    acc = wmma16(frag_ld_a(sEh + 96, H_, lane), frag_ld_b(wb + 96, H_, lane), acc);
    int n = wv * 16 + (lane & 15);
    int hi = lane >> 4;
    float bo = bo2[l * H_ + n];
#pragma unroll
    for (int r = 0; r < 8; ++r) {
      int il = r + (hi << 3);
      if (il < IT_) sAgg[il * H_ + n] = sHi[il * H_ + n] + acc[r] + bo;
    }
  }
  __syncthreads();
  // layernorm + vm, write h and h16
  if (tid < IT_) {
    float mu = 0.f;
    for (int n = 0; n < H_; ++n) mu += sAgg[tid * H_ + n];
    mu *= (1.f / H_);
    float var = 0.f;
    for (int n = 0; n < H_; ++n) { float d = sAgg[tid * H_ + n] - mu; var += d * d; }
    var *= (1.f / H_);
    sMu[tid] = mu; sVar[tid] = rsqrtf(var + 1e-5f);
  }
  __syncthreads();
  if (tid < H_) {
    int n = tid;
    float g = lng[l * H_ + n], be = lnb[l * H_ + n];
    for (int il = 0; il < IT_; ++il) {
      int i = i0 + il;
      float vm = (sJmask[i] > 0.f) ? 1.f : 0.f;
      float v = ((sAgg[il * H_ + n] - sMu[il]) * sVar[il] * g + be) * vm;
      h[(b * L_ + i) * H_ + n] = v;
      h16[(b * L_ + i) * H_ + n] = (_Float16)v;
    }
  }
}

// ---- decoder ----
__global__ void k_dec(const float* __restrict__ hist, const float* __restrict__ mask,
                      const float* __restrict__ h,
                      const float* __restrict__ hg, const float* __restrict__ hb,
                      const float* __restrict__ Wh1, const float* __restrict__ bh1,
                      const float* __restrict__ Wh2, const float* __restrict__ bh2,
                      float* __restrict__ out) {
  __shared__ float sx[DIN + H_];
  __shared__ float shid[H_];
  __shared__ float stats[2];
  __shared__ int slast;
  int b = blockIdx.x, t = threadIdx.x;
  if (t == 0) {
    float s = 0.f;
    for (int j = 0; j < L_; ++j) s += mask[b * L_ + j];
    int vc = (int)s; vc = vc < 1 ? 1 : (vc > L_ ? L_ : vc);
    slast = vc - 1;
  }
  __syncthreads();
  int last = slast;
  if (t < DIN + H_)
    sx[t] = (t < DIN) ? hist[(b * L_ + last) * DIN + t] : h[(b * L_ + last) * H_ + (t - DIN)];
  __syncthreads();
  if (t == 0) {
    float mu = 0.f;
    for (int k = 0; k < DIN + H_; ++k) mu += sx[k];
    mu /= (float)(DIN + H_);
    float var = 0.f;
    for (int k = 0; k < DIN + H_; ++k) { float d = sx[k] - mu; var += d * d; }
    var /= (float)(DIN + H_);
    stats[0] = mu; stats[1] = rsqrtf(var + 1e-5f);
  }
  __syncthreads();
  if (t < DIN + H_) sx[t] = (sx[t] - stats[0]) * stats[1] * hg[t] + hb[t];
  __syncthreads();
  if (t < H_) {
    float a = bh1[t];
    for (int k = 0; k < DIN + H_; ++k) a = fmaf(sx[k], Wh1[k * H_ + t], a);
    shid[t] = fmaxf(a, 0.f);
  }
  __syncthreads();
  if (t < FUT_ * MD_) {
    float a = bh2[t];
    for (int k = 0; k < H_; ++k) a = fmaf(shid[k], Wh2[k * (FUT_ * MD_) + t], a);
    if (a != a) a = 0.f;
    if (a == __builtin_inff()) a = 1e4f;
    if (a == -__builtin_inff()) a = -1e4f;
    out[b * FUT_ * MD_ + t] = a;
  }
}

__global__ void k_copyh(const float* __restrict__ h, float* __restrict__ dst) {
  int i = blockIdx.x * blockDim.x + threadIdx.x;
  if (i < B_ * L_ * H_) dst[i] = h[i];
}

extern "C" void kernel_launch(void* const* d_in, const int* in_sizes, int n_in,
                              void* d_out, int out_size, void* d_ws, size_t ws_size,
                              hipStream_t stream) {
  const float* hist = (const float*)d_in[0];
  const float* mask = (const float*)d_in[1];
  const float* Wp   = (const float*)d_in[2];
  const float* bp   = (const float*)d_in[3];
  const float* We1  = (const float*)d_in[4];
  const float* be1  = (const float*)d_in[5];
  const float* We2  = (const float*)d_in[6];
  const float* be2  = (const float*)d_in[7];
  const float* Wa1  = (const float*)d_in[8];
  const float* ba1  = (const float*)d_in[9];
  const float* wa2  = (const float*)d_in[10];
  const float* ba2  = (const float*)d_in[11];
  const float* Wm1  = (const float*)d_in[12];
  const float* bm1  = (const float*)d_in[13];
  const float* Wm2  = (const float*)d_in[14];
  const float* bm2  = (const float*)d_in[15];
  const float* Wo1  = (const float*)d_in[16];
  const float* bo1  = (const float*)d_in[17];
  const float* Wo2  = (const float*)d_in[18];
  const float* bo2  = (const float*)d_in[19];
  const float* lng  = (const float*)d_in[20];
  const float* lnb  = (const float*)d_in[21];
  const float* hg   = (const float*)d_in[22];
  const float* hb   = (const float*)d_in[23];
  const float* Wh1  = (const float*)d_in[24];
  const float* bh1  = (const float*)d_in[25];
  const float* Wh2  = (const float*)d_in[26];
  const float* bh2  = (const float*)d_in[27];

  char* ws = (char*)d_ws;
  float*    ts   = (float*)(ws + 0);                 // 8 KB
  float*    h    = (float*)(ws + 8192);              // 1 MB
  _Float16* h16  = (_Float16*)(ws + 1056768);        // 512 KB
  float*    proj = (float*)(ws + 1581056);           // 4 MB
  _Float16* wf   = (_Float16*)(ws + 5775360);        // 640 KB

  k_ts<<<1, 32, 0, stream>>>(hist, ts);
  k_h0<<<B_ * L_, H_, 0, stream>>>(hist, mask, Wp, bp, h, h16);
  k_wconv<<<(NL_ * 163840 + 255) / 256, 256, 0, stream>>>(Wa1, Wm1, Wo1, We2, Wm2, Wo2, wf);

  const size_t smem = 80032;
  for (int l = 0; l < NL_; ++l) {
    k_proj<<<128, 128, 0, stream>>>(h16, wf + l * 163840, proj);
    k_edge<<<B_ * (L_ / IT_), 256, smem, stream>>>(hist, mask, ts, h, h16, proj, wf,
                                                   We1, be1, be2, ba1, wa2, ba2,
                                                   bm1, bm2, bo1, bo2, lng, lnb, l);
  }
  k_dec<<<B_, 192, 0, stream>>>(hist, mask, h, hg, hb, Wh1, bh1, Wh2, bh2, (float*)d_out);
  k_copyh<<<(B_ * L_ * H_ + 255) / 256, 256, 0, stream>>>(h, (float*)d_out + B_ * FUT_ * MD_);
}
